// GCN_89859305766966
// MI455X (gfx1250) — matrix operand order repaired and checked by reading, not compiled, not run on hardware
//
#include <hip/hip_runtime.h>
#include <hip/hip_bf16.h>

// ---------------------------------------------------------------------------
// 2-layer GCN for MI455X (gfx1250, wave32).
//   layer: agg[d] += (x@W)[s] * dinv[s]*dinv[d]  (edges, f32 atomics in L2)
//          h = act(agg + (x@W)*dinv^2 + b)
// GEMMs run on v_wmma_f32_16x16x32_bf16 with pre-converted bf16 weights;
// the edge scatter phase is the memory-bound part (gathers + f32 atomics
// resolve mostly in the 192MB L2 since xw/agg buffers are L2-resident).
// ---------------------------------------------------------------------------

typedef __attribute__((ext_vector_type(16))) __bf16 v16bf;
typedef __attribute__((ext_vector_type(8)))  float  v8f;

// ---------------- degree / normalization ----------------------------------
__global__ void gcn_deg_init(float* __restrict__ deg, int M) {
    int i = blockIdx.x * blockDim.x + threadIdx.x;
    if (i < M) deg[i] = 1.0f;                       // self-loop
}

__global__ void gcn_deg_count(const int* __restrict__ dst, float* __restrict__ deg, int E) {
    int e = blockIdx.x * blockDim.x + threadIdx.x;
    if (e < E) atomicAdd(&deg[dst[e]], 1.0f);
}

__global__ void gcn_deg_rsqrt(float* __restrict__ deg, int M) {
    int i = blockIdx.x * blockDim.x + threadIdx.x;
    if (i < M) deg[i] = rsqrtf(deg[i]);             // deg >= 1 always
}

// ---------------- weight pre-conversion: W[128,N] f32 -> Wt[N,128] bf16 ----
// Wt is K-contiguous so a B fragment (two 8-K runs for one column) is
// two 16-byte global loads, no conversion in the GEMM hot loop.
__global__ void gcn_wt_bf16(const float* __restrict__ W, __bf16* __restrict__ Wt, int N) {
    int idx = blockIdx.x * blockDim.x + threadIdx.x;   // over N*128
    if (idx >= N * 128) return;
    int n = idx >> 7, k = idx & 127;
    Wt[n * 128 + k] = (__bf16)W[k * N + n];
}

// ---------------- WMMA GEMM: OUT[M,N] = X[M,128] @ W[128,N] ----------------
// Block = RT row-tiles of 16 rows; wave = one 16-column tile (N/16 waves).
// A (16x32 bf16) ISA layout: lane m=lane&15; halves 0-7 -> K=kb..kb+7,
//   halves 8-15 -> K=kb+16..kb+23, kb=(lane<16?0:8)  => two b128 LDS loads.
// B (32x16 bf16): lane n=lane&15; same K order as A (kb..kb+7, kb+16..kb+23)
//   => two 8-bf16 runs from Wt => two b128 global loads, kept in VGPRs
//   for all 4 K-chunks and reused across the RT row tiles.
// D (16x16 f32): lane n=lane&15; VGPR r -> row (lane<16?0:8)+r.
template <int N, int RT>
__global__ void __launch_bounds__(32 * (N / 16))
gcn_gemm_wmma(const float* __restrict__ X, const __bf16* __restrict__ Wt,
              float* __restrict__ OUT, int M) {
    constexpr int K = 128;
    __shared__ __align__(16) __bf16 sA[RT * 16 * K];   // bf16 X-tiles

    const int m0 = blockIdx.x * (RT * 16);

    for (int idx = threadIdx.x; idx < RT * 16 * K; idx += blockDim.x) {
        const int r = idx >> 7, c = idx & (K - 1);
        const int row = m0 + r;
        const float v = (row < M) ? X[(size_t)row * K + c] : 0.0f;
        sA[idx] = (__bf16)v;
    }
    __syncthreads();

    const int lane = threadIdx.x & 31;
    const int wave = threadIdx.x >> 5;
    const int n0   = wave * 16;
    const int nn   = lane & 15;
    const int hi   = lane >> 4;       // half-wave select
    const int kbA  = hi * 8;
    const int kbB  = hi * 8;

    // All 4 B-chunk fragments, loaded once, reused for every row tile.
    union Frag { v16bf v; uint4 q[2]; };
    Frag b[4];
    const __bf16* wrow = Wt + (size_t)(n0 + nn) * K;
#pragma unroll
    for (int c = 0; c < 4; ++c) {
        const __bf16* bp = wrow + c * 32 + kbB;
        b[c].q[0] = *(const uint4*)(bp);       // K = kbB .. +7
        b[c].q[1] = *(const uint4*)(bp + 16);  // K = kbB+16 .. +23
    }

#pragma unroll
    for (int t = 0; t < RT; ++t) {
        v8f acc = {};
#pragma unroll
        for (int c = 0; c < 4; ++c) {
            Frag a;
            const __bf16* ap = &sA[(t * 16 + nn) * K + c * 32 + kbA];
            a.q[0] = *(const uint4*)(ap);        // K = kc+kbA .. +7
            a.q[1] = *(const uint4*)(ap + 16);   // K = kc+kbA+16 .. +23
            acc = __builtin_amdgcn_wmma_f32_16x16x32_bf16(
                /*neg_a=*/false, a.v, /*neg_b=*/false, b[c].v,
                /*c_mod=*/(short)0, acc, /*reuse_a=*/false, /*reuse_b=*/false);
        }

        const int mt = m0 + t * 16;
        float* orow = OUT + (size_t)(mt + hi * 8) * N + (n0 + nn);
        if (mt + 16 <= M) {                      // full tile: branch-free stores
#pragma unroll
            for (int r = 0; r < 8; ++r) orow[(size_t)r * N] = acc[r];
        } else if (mt < M) {                     // generic tail (unused for M=100000)
#pragma unroll
            for (int r = 0; r < 8; ++r)
                if (mt + hi * 8 + r < M) orow[(size_t)r * N] = acc[r];
        }
    }
}

// ---------------- edge scatter (one wave per edge) -------------------------
__global__ void gcn_scatter128(const float* __restrict__ xw, const float* __restrict__ dinv,
                               const int* __restrict__ src, const int* __restrict__ dst,
                               float* agg, int E) {
    const int lane = threadIdx.x & 31;
    const int e = blockIdx.x * (blockDim.x >> 5) + (threadIdx.x >> 5);
    if (e >= E) return;
    const int s = src[e], d = dst[e];
    const float nrm = dinv[s] * dinv[d];
    const float4 v = ((const float4*)(xw + (size_t)s * 128))[lane];
    float* base = agg + (size_t)d * 128 + lane * 4;
    atomicAdd(base + 0, v.x * nrm);
    atomicAdd(base + 1, v.y * nrm);
    atomicAdd(base + 2, v.z * nrm);
    atomicAdd(base + 3, v.w * nrm);
}

__global__ void gcn_scatter64(const float* __restrict__ xw, const float* __restrict__ dinv,
                              const int* __restrict__ src, const int* __restrict__ dst,
                              float* agg, int E) {
    const int lane = threadIdx.x & 31;
    const int e = blockIdx.x * (blockDim.x >> 5) + (threadIdx.x >> 5);
    if (e >= E) return;
    const int s = src[e], d = dst[e];
    const float nrm = dinv[s] * dinv[d];
    const float2 v = ((const float2*)(xw + (size_t)s * 64))[lane];
    float* base = agg + (size_t)d * 64 + lane * 2;
    atomicAdd(base + 0, v.x * nrm);
    atomicAdd(base + 1, v.y * nrm);
}

// ---------------- epilogue 1: self-loop + bias + ReLU (in place) -----------
__global__ void gcn_epilogue1_relu(float* hx /*in: agg1, out: h*/,
                                   const float* __restrict__ xw,
                                   const float* __restrict__ dinv,
                                   const float* __restrict__ b, int M) {
    const int idx = blockIdx.x * blockDim.x + threadIdx.x;   // 4 floats/thread
    if (idx >= M * 32) return;
    const int row = idx >> 5;
    const int c4  = (idx & 31) * 4;
    const float dd = dinv[row];
    const float selfn = dd * dd;
    const size_t base = (size_t)row * 128 + c4;
    const float4 a  = *(const float4*)(hx + base);
    const float4 w  = *(const float4*)(xw + base);
    const float4 bb = *(const float4*)(b + c4);
    float4 r;
    r.x = fmaxf(a.x + w.x * selfn + bb.x, 0.0f);
    r.y = fmaxf(a.y + w.y * selfn + bb.y, 0.0f);
    r.z = fmaxf(a.z + w.z * selfn + bb.z, 0.0f);
    r.w = fmaxf(a.w + w.w * selfn + bb.w, 0.0f);
    *(float4*)(hx + base) = r;
}

// ---------------- epilogue 2: self-loop + bias + log_softmax (64 cols) -----
__global__ void gcn_epilogue2_lsm(const float* __restrict__ agg2,
                                  const float* __restrict__ xw2,
                                  const float* __restrict__ dinv,
                                  const float* __restrict__ b2,
                                  float* __restrict__ out, int M) {
    const int lane = threadIdx.x & 31;                       // wave32: 2 cols/lane
    const int row = blockIdx.x * (blockDim.x >> 5) + (threadIdx.x >> 5);
    if (row >= M) return;
    const float dd = dinv[row];
    const float selfn = dd * dd;
    const int f0 = lane * 2;
    const size_t base = (size_t)row * 64 + f0;
    float vx = agg2[base]     + xw2[base]     * selfn + b2[f0];
    float vy = agg2[base + 1] + xw2[base + 1] * selfn + b2[f0 + 1];

    float mx = fmaxf(vx, vy);
#pragma unroll
    for (int m = 16; m >= 1; m >>= 1) mx = fmaxf(mx, __shfl_xor(mx, m, 32));
    float s = expf(vx - mx) + expf(vy - mx);
#pragma unroll
    for (int m = 16; m >= 1; m >>= 1) s += __shfl_xor(s, m, 32);
    const float lse = mx + logf(s);
    out[base]     = vx - lse;
    out[base + 1] = vy - lse;
}

// ---------------------------------------------------------------------------
extern "C" void kernel_launch(void* const* d_in, const int* in_sizes, int n_in,
                              void* d_out, int out_size, void* d_ws, size_t ws_size,
                              hipStream_t stream) {
    const float* x  = (const float*)d_in[0];
    const int*   ei = (const int*)d_in[1];     // edge_index [2, E] flat
    const float* W1 = (const float*)d_in[2];
    const float* b1 = (const float*)d_in[3];
    const float* W2 = (const float*)d_in[4];
    const float* b2 = (const float*)d_in[5];
    float* out = (float*)d_out;

    const int M = in_sizes[0] / 128;   // 100000
    const int E = in_sizes[1] / 2;     // 1600000
    const int* src = ei;
    const int* dst = ei + E;

    char* ws = (char*)d_ws;
    size_t off = 0;
    auto wsalloc = [&](size_t bytes) -> void* {
        void* p = ws + off;
        off += (bytes + 255) & ~(size_t)255;
        return p;
    };
    float*  dinv = (float*)wsalloc((size_t)M * 4);
    float*  xw1  = (float*)wsalloc((size_t)M * 128 * 4);
    float*  agg1 = (float*)wsalloc((size_t)M * 128 * 4);   // becomes h in place
    float*  xw2  = (float*)wsalloc((size_t)M * 64 * 4);
    float*  agg2 = (float*)wsalloc((size_t)M * 64 * 4);
    __bf16* Wt1  = (__bf16*)wsalloc((size_t)128 * 128 * 2);
    __bf16* Wt2  = (__bf16*)wsalloc((size_t)64 * 128 * 2);
    (void)ws_size; (void)n_in; (void)out_size;

    // ---- normalization coefficients + weight conversion ----
    gcn_deg_init <<<(M + 255) / 256, 256, 0, stream>>>(dinv, M);
    gcn_deg_count<<<(E + 255) / 256, 256, 0, stream>>>(dst, dinv, E);
    gcn_deg_rsqrt<<<(M + 255) / 256, 256, 0, stream>>>(dinv, M);
    gcn_wt_bf16<<<(128 * 128 + 255) / 256, 256, 0, stream>>>(W1, Wt1, 128);
    gcn_wt_bf16<<<(64 * 128 + 255) / 256, 256, 0, stream>>>(W2, Wt2, 64);

    constexpr int RT = 4;                        // row tiles per GEMM block
    const int gemmGrid = (M + RT * 16 - 1) / (RT * 16);

    // ---- layer 1: 128 -> 128 ----
    gcn_gemm_wmma<128, RT><<<gemmGrid, 256, 0, stream>>>(x, Wt1, xw1, M);
    hipMemsetAsync(agg1, 0, (size_t)M * 128 * 4, stream);
    gcn_scatter128<<<(E + 7) / 8, 256, 0, stream>>>(xw1, dinv, src, dst, agg1, E);
    gcn_epilogue1_relu<<<(M * 32 + 255) / 256, 256, 0, stream>>>(agg1, xw1, dinv, b1, M);

    // ---- layer 2: 128 -> 64 ----
    gcn_gemm_wmma<64, RT><<<gemmGrid, 128, 0, stream>>>(agg1, Wt2, xw2, M);
    hipMemsetAsync(agg2, 0, (size_t)M * 64 * 4, stream);
    gcn_scatter64<<<(E + 7) / 8, 256, 0, stream>>>(xw2, dinv, src, dst, agg2, E);
    gcn_epilogue2_lsm<<<(M + 7) / 8, 256, 0, stream>>>(agg2, xw2, dinv, b2, out, M);
}
